// SparseMoE_24043226923402
// MI455X (gfx1250) — matrix operand order repaired and checked by reading, not compiled, not run on hardware
//
#include <hip/hip_runtime.h>
#include <hip/hip_bf16.h>
#include <math.h>

// ---------------------------------------------------------------------------
// SparseMoE forward for MI455X (gfx1250, wave32, WMMA + TDM)
//   B=4 T=1024 D=1024 E=8 H=4096 K=2, N = 4096 tokens
// Routing-sparse expert FFN, bf16 WMMA 16x16x32 (f32 accum), fp32 elsewhere.
// A-panels staged LDS via Tensor Data Mover (gather mode for token lists).
// ---------------------------------------------------------------------------

#define NTOK 4096
#define DDIM 1024
#define HDIM 4096
#define NEXP 8
#define KTOP 2
#define EPS_LN 1e-5f
#define EPS_W  1e-8f
#define BAL_W  0.01f
#define ROWS   64   // token rows per FFN block tile

typedef __attribute__((ext_vector_type(16))) __bf16 v16bf;
typedef __attribute__((ext_vector_type(8)))  float  v8f;
typedef __attribute__((ext_vector_type(4)))  unsigned int u32x4;
typedef __attribute__((ext_vector_type(8)))  int i32x8;
typedef __attribute__((ext_vector_type(4)))  int i32x4;

#if defined(__has_builtin)
#if __has_builtin(__builtin_amdgcn_tensor_load_to_lds)
#define USE_TDM 1
#else
#define USE_TDM 0
#endif
#else
#define USE_TDM 0
#endif

#if USE_TDM
__device__ inline void tdm_load(u32x4 g0, i32x8 g1, i32x4 g2, i32x4 g3) {
#if __clang_major__ >= 23
  i32x8 z8 = {0, 0, 0, 0, 0, 0, 0, 0};
  __builtin_amdgcn_tensor_load_to_lds(g0, g1, g2, g3, z8, 0);
#else
  __builtin_amdgcn_tensor_load_to_lds(g0, g1, g2, g3, 0);
#endif
}

// Build D# group0: count=1, lds byte addr, 57-bit global addr, type=2.
__device__ inline u32x4 tdm_group0(unsigned lds_addr, const void* gaddr, bool gather) {
  unsigned long long ga = (unsigned long long)(uintptr_t)gaddr;
  u32x4 g0;
  g0[0] = 1u | (gather ? 0x80000000u : 0u);  // count=1, gather_mode bit31, 16-bit idx
  g0[1] = lds_addr;
  g0[2] = (unsigned)(ga & 0xFFFFFFFFu);
  g0[3] = (unsigned)((ga >> 32) & 0x01FFFFFFu) | (2u << 30);  // type=2 ("image")
  return g0;
}

// Build D# group1: data_size=2B, dims/strides (elements), tile dims.
__device__ inline i32x8 tdm_group1(unsigned dim0, unsigned dim1,
                                   unsigned tile0, unsigned tile1,
                                   unsigned long long stride0) {
  i32x8 g1;
  g1[0] = (1 << 16);                                   // wg_mask=0, data_size=1 (2B)
  g1[1] = (int)((dim0 & 0xFFFFu) << 16);               // tensor_dim0[15:0] @ bits63:48
  g1[2] = (int)((dim0 >> 16) | ((dim1 & 0xFFFFu) << 16));
  g1[3] = (int)((dim1 >> 16) | ((tile0 & 0xFFFFu) << 16));
  g1[4] = (int)(tile1 & 0xFFFFu);                      // tile_dim1, tile_dim2=0
  g1[5] = (int)(stride0 & 0xFFFFFFFFu);                // tensor_dim0_stride[31:0]
  g1[6] = (int)((stride0 >> 32) & 0xFFFFu);            // stride hi; dim1_stride=0
  g1[7] = 0;
  return g1;
}
#endif  // USE_TDM

// ---------------------------------------------------------------------------
// LayerNorm: one block per token row. Writes f32 (router) and bf16 (FFN A).
// ---------------------------------------------------------------------------
__global__ __launch_bounds__(256) void ln_kernel(const float* __restrict__ x,
                                                 const float* __restrict__ gamma,
                                                 const float* __restrict__ beta,
                                                 float* __restrict__ xn,
                                                 __bf16* __restrict__ xnb) {
  const int row = blockIdx.x;
  const int tid = threadIdx.x;
  const size_t base = (size_t)row * DDIM + tid * 4;
  float4 v = *(const float4*)(x + base);
  float s  = v.x + v.y + v.z + v.w;
  float ss = v.x * v.x + v.y * v.y + v.z * v.z + v.w * v.w;
  __shared__ float r1[256], r2[256];
  r1[tid] = s; r2[tid] = ss;
  __syncthreads();
  for (int st = 128; st > 0; st >>= 1) {
    if (tid < st) { r1[tid] += r1[tid + st]; r2[tid] += r2[tid + st]; }
    __syncthreads();
  }
  const float mu   = r1[0] * (1.0f / DDIM);
  const float var  = r2[0] * (1.0f / DDIM) - mu * mu;
  const float rstd = rsqrtf(var + EPS_LN);
  float4 g  = *(const float4*)(gamma + tid * 4);
  float4 bb = *(const float4*)(beta + tid * 4);
  float4 o;
  o.x = (v.x - mu) * rstd * g.x + bb.x;
  o.y = (v.y - mu) * rstd * g.y + bb.y;
  o.z = (v.z - mu) * rstd * g.z + bb.z;
  o.w = (v.w - mu) * rstd * g.w + bb.w;
  *(float4*)(xn + base) = o;
  union { __bf16 h[4]; uint2 q; } u;
  u.h[0] = (__bf16)o.x; u.h[1] = (__bf16)o.y; u.h[2] = (__bf16)o.z; u.h[3] = (__bf16)o.w;
  *(uint2*)(xnb + base) = u.q;
}

// ---------------------------------------------------------------------------
// Router: one block per token. logits -> softmax -> top2 -> expert lists.
// ---------------------------------------------------------------------------
__global__ __launch_bounds__(256) void router_kernel(const float* __restrict__ xn,
                                                     const float* __restrict__ gate_w,
                                                     float* __restrict__ probs_out,
                                                     int* __restrict__ cnt,
                                                     int* __restrict__ toklist,
                                                     float* __restrict__ gwlist,
                                                     int* __restrict__ klist) {
  const int nrow = blockIdx.x;
  const int tid  = threadIdx.x;
  float acc[NEXP];
#pragma unroll
  for (int e = 0; e < NEXP; ++e) acc[e] = 0.0f;
  for (int d = tid; d < DDIM; d += 256) {
    const float xv = xn[(size_t)nrow * DDIM + d];
    const float* g = gate_w + (size_t)d * NEXP;
#pragma unroll
    for (int e = 0; e < NEXP; ++e) acc[e] += xv * g[e];
  }
  __shared__ float red[NEXP * 256];
#pragma unroll
  for (int e = 0; e < NEXP; ++e) red[e * 256 + tid] = acc[e];
  __syncthreads();
  for (int st = 128; st > 0; st >>= 1) {
    if (tid < st) {
#pragma unroll
      for (int e = 0; e < NEXP; ++e) red[e * 256 + tid] += red[e * 256 + tid + st];
    }
    __syncthreads();
  }
  if (tid == 0) {
    float l[NEXP], m = -1e30f;
#pragma unroll
    for (int e = 0; e < NEXP; ++e) { l[e] = red[e * 256]; m = fmaxf(m, l[e]); }
    float p[NEXP], sum = 0.0f;
#pragma unroll
    for (int e = 0; e < NEXP; ++e) { p[e] = __expf(l[e] - m); sum += p[e]; }
    const float inv = 1.0f / sum;
#pragma unroll
    for (int e = 0; e < NEXP; ++e) {
      p[e] *= inv;
      probs_out[(size_t)nrow * NEXP + e] = p[e];
    }
    int e0 = 0;
#pragma unroll
    for (int e = 1; e < NEXP; ++e) if (p[e] > p[e0]) e0 = e;
    int e1 = (e0 == 0) ? 1 : 0;
#pragma unroll
    for (int e = 0; e < NEXP; ++e) if (e != e0 && p[e] > p[e1]) e1 = e;
    const float norm = 1.0f / (p[e0] + p[e1] + EPS_W);
    const float w0 = p[e0] * norm, w1 = p[e1] * norm;
    int s0 = atomicAdd(&cnt[e0], 1);
    toklist[e0 * NTOK + s0] = nrow; gwlist[e0 * NTOK + s0] = w0; klist[e0 * NTOK + s0] = 0;
    int s1 = atomicAdd(&cnt[e1], 1);
    toklist[e1 * NTOK + s1] = nrow; gwlist[e1 * NTOK + s1] = w1; klist[e1 * NTOK + s1] = 1;
  }
}

// ---------------------------------------------------------------------------
// Deterministic mean(router_probs) per expert; offsets; balance loss.
// ---------------------------------------------------------------------------
__global__ __launch_bounds__(256) void probs_reduce_kernel(const float* __restrict__ probs,
                                                           float* __restrict__ probsSum) {
  const int e = blockIdx.x, tid = threadIdx.x;
  float s = 0.0f;
  for (int n = tid; n < NTOK; n += 256) s += probs[(size_t)n * NEXP + e];
  __shared__ float r[256];
  r[tid] = s;
  __syncthreads();
  for (int st = 128; st > 0; st >>= 1) {
    if (tid < st) r[tid] += r[tid + st];
    __syncthreads();
  }
  if (tid == 0) probsSum[e] = r[0];
}

__global__ void offsets_kernel(const int* __restrict__ cnt, int* __restrict__ offs) {
  int o = 0;
  for (int e = 0; e < NEXP; ++e) { offs[e] = o; o += cnt[e]; }
  offs[NEXP] = o;
}

__global__ void loss_kernel(const int* __restrict__ cnt, const float* __restrict__ probsSum,
                            float* __restrict__ loss_out) {
  float loss = 0.0f;
  for (int e = 0; e < NEXP; ++e)
    loss += ((float)cnt[e] / (float)NTOK) * (probsSum[e] / (float)NTOK);
  loss_out[0] = BAL_W * (float)NEXP * loss;
}

// ---------------------------------------------------------------------------
// Prepack f32 weights (E, Kdim, Nc) -> bf16 WMMA B-fragment-linear layout.
// B 32x16 (contiguous k, same order as A): col = lane%16; lanes 0-15 K=0..15, lanes 16-31 K=16..31.
// ---------------------------------------------------------------------------
__global__ __launch_bounds__(256) void prepack_kernel(const float* __restrict__ src,
                                                      __bf16* __restrict__ dst,
                                                      int Kdim, int Nc) {
  const int idx  = blockIdx.x * 256 + threadIdx.x;
  const int lane = idx & 31;
  const int blk  = idx >> 5;
  const int ksteps = Kdim >> 5;
  const int ks = blk % ksteps;
  const int t  = blk / ksteps;
  const int nt = t % (Nc >> 4);
  const int e  = t / (Nc >> 4);
  if (e >= NEXP) return;
  const int kadd = (lane & 16) ? 16 : 0;
  const float* s = src + ((size_t)e * Kdim + (size_t)ks * 32 + kadd) * Nc + nt * 16 + (lane & 15);
  union { __bf16 hh[16]; uint4 q[2]; } u;
#pragma unroll
  for (int j = 0; j < 16; ++j) u.hh[j] = (__bf16)s[(size_t)j * Nc];
  __bf16* dp = dst + (size_t)blk * 512 + lane * 16;
  *(uint4*)(dp)     = u.q[0];
  *(uint4*)(dp + 8) = u.q[1];
}

// ---------------------------------------------------------------------------
// Fragment loaders.
// A 16x32: lane<16 -> M=lane, K 0..15; lane>=16 -> K 16..31.
// ---------------------------------------------------------------------------
__device__ inline v16bf load_a_frag(const __bf16* panel, int lane, int ks) {
  const int m  = lane & 15;
  const int kb = ks * 32 + ((lane & 16) ? 16 : 0);
  union { v16bf v; uint4 q[2]; } u;
  u.q[0] = *(const uint4*)(panel + m * 512 + kb);
  u.q[1] = *(const uint4*)(panel + m * 512 + kb + 8);
  return u.v;
}

__device__ inline v16bf load_b_frag(const __bf16* wf, size_t bblk, int lane) {
  union { v16bf v; uint4 q[2]; } u;
  const __bf16* p = wf + bblk * 512 + lane * 16;
  u.q[0] = *(const uint4*)(p);
  u.q[1] = *(const uint4*)(p + 8);
  return u.v;
}

// ---------------------------------------------------------------------------
// FFN1: h = gelu(Xg @ w1[e] + b1[e]) for routed tokens of expert e.
// Block = 64 tokens x 128 H cols; 8 waves; per-wave 4 C tiles (B reused 4x).
// A panel staged via TDM gather (token-id row indices), 16 rows/descriptor.
// ---------------------------------------------------------------------------
__global__ __launch_bounds__(256) void ffn1_kernel(const __bf16* __restrict__ xnb,
                                                   const __bf16* __restrict__ w1f,
                                                   const float* __restrict__ b1,
                                                   __bf16* __restrict__ h,
                                                   const int* __restrict__ toklist,
                                                   const int* __restrict__ cnt,
                                                   const int* __restrict__ offs) {
  const int e = blockIdx.z;
  const int c = cnt[e];
  const int slot0 = blockIdx.y * ROWS;
  if (slot0 >= c) return;
  const int tid = threadIdx.x, lane = tid & 31, wv = tid >> 5;
  const int ht16 = blockIdx.x * 8 + wv;  // 16-col tile index in H

  __shared__ __align__(16) __bf16 apanel[ROWS * 512];
  __shared__ int toks[ROWS];
  if (tid < ROWS) {
    const int s = slot0 + tid;
    toks[tid] = (s < c) ? toklist[e * NTOK + s] : -1;
  }

  v8f acc[4];
#pragma unroll
  for (int mt = 0; mt < 4; ++mt) acc[mt] = (v8f){0.f,0.f,0.f,0.f,0.f,0.f,0.f,0.f};

  for (int kc = 0; kc < DDIM / 512; ++kc) {
    __syncthreads();
#if USE_TDM
    if (wv == 0) {
      const unsigned lds0 = (unsigned)(uintptr_t)(&apanel[0]);
#pragma unroll
      for (int dsc = 0; dsc < 4; ++dsc) {
        const int r0 = dsc * 16;
        const int valid = min(16, c - (slot0 + r0));
        if (valid > 0) {
          i32x4 g2, g3;
#pragma unroll
          for (int w = 0; w < 8; ++w) {
            int t0 = toks[r0 + w * 2];     if (t0 < 0) t0 = 0;
            int t1 = toks[r0 + w * 2 + 1]; if (t1 < 0) t1 = 0;
            const int packed = (t0 & 0xFFFF) | ((t1 & 0xFFFF) << 16);
            if (w < 4) g2[w] = packed; else g3[w - 4] = packed;
          }
          tdm_load(tdm_group0(lds0 + r0 * 512 * 2, xnb + kc * 512, /*gather=*/true),
                   tdm_group1(DDIM, NTOK, 512, (unsigned)valid, DDIM),
                   g2, g3);
        }
      }
      __builtin_amdgcn_s_wait_tensorcnt(0);
    }
#else
    for (int i = tid; i < ROWS * 64; i += 256) {
      const int r = i >> 6, c8 = i & 63;
      uint4 val = {0, 0, 0, 0};
      const int t = toks[r];
      if (t >= 0) val = *(const uint4*)(xnb + (size_t)t * DDIM + kc * 512 + c8 * 8);
      *(uint4*)(&apanel[r * 512 + c8 * 8]) = val;
    }
#endif
    __syncthreads();
#pragma unroll
    for (int ks = 0; ks < 16; ++ks) {
      const size_t bblk = (size_t)(e * (HDIM / 16) + ht16) * (DDIM / 32) + (kc * 16 + ks);
      const v16bf b = load_b_frag(w1f, bblk, lane);
#pragma unroll
      for (int mt = 0; mt < 4; ++mt) {
        const v16bf a = load_a_frag(apanel + mt * 16 * 512, lane, ks);
        acc[mt] = __builtin_amdgcn_wmma_f32_16x16x32_bf16(false, a, false, b, (short)0,
                                                          acc[mt], false, false);
      }
    }
  }
  // GELU (exact erf) + store bf16 to h rows [offs[e]+slot]
  const int n = lane & 15;
  const int radd = (lane & 16) ? 8 : 0;
  const float bias = b1[e * HDIM + ht16 * 16 + n];
  const int off = offs[e];
#pragma unroll
  for (int mt = 0; mt < 4; ++mt) {
#pragma unroll
    for (int r = 0; r < 8; ++r) {
      const int m = mt * 16 + r + radd;
      const int s = slot0 + m;
      if (s < c) {
        const float xv = acc[mt][r] + bias;
        const float gl = 0.5f * xv * (1.0f + erff(xv * 0.70710678118654752f));
        h[(size_t)(off + s) * HDIM + ht16 * 16 + n] = (__bf16)gl;
      }
    }
  }
}

// ---------------------------------------------------------------------------
// FFN2: contrib[(tok*2+k)] = gate_w * (h @ w2[e] + b2[e]) for routed tokens.
// Rows are contiguous in h -> plain 2D TDM tile load per chunk.
// ---------------------------------------------------------------------------
__global__ __launch_bounds__(256) void ffn2_kernel(const __bf16* __restrict__ h,
                                                   const __bf16* __restrict__ w2f,
                                                   const float* __restrict__ b2,
                                                   float* __restrict__ contrib,
                                                   const int* __restrict__ toklist,
                                                   const float* __restrict__ gwlist,
                                                   const int* __restrict__ klist,
                                                   const int* __restrict__ cnt,
                                                   const int* __restrict__ offs) {
  const int e = blockIdx.z;
  const int c = cnt[e];
  const int slot0 = blockIdx.y * ROWS;
  if (slot0 >= c) return;
  const int tid = threadIdx.x, lane = tid & 31, wv = tid >> 5;
  const int dt16 = blockIdx.x * 8 + wv;  // 16-col tile index in D
  const int off = offs[e];
  const int valid = min(ROWS, c - slot0);

  __shared__ __align__(16) __bf16 apanel[ROWS * 512];
  __shared__ int toks[ROWS];
  __shared__ float gws[ROWS];
  __shared__ int kids[ROWS];
  if (tid < ROWS) {
    const int s = slot0 + tid;
    if (s < c) {
      toks[tid] = toklist[e * NTOK + s];
      gws[tid]  = gwlist[e * NTOK + s];
      kids[tid] = klist[e * NTOK + s];
    } else {
      toks[tid] = 0; gws[tid] = 0.0f; kids[tid] = 0;
    }
  }

  v8f acc[4];
#pragma unroll
  for (int mt = 0; mt < 4; ++mt) acc[mt] = (v8f){0.f,0.f,0.f,0.f,0.f,0.f,0.f,0.f};

  for (int kc = 0; kc < HDIM / 512; ++kc) {
    __syncthreads();
#if USE_TDM
    if (wv == 0) {
      const unsigned lds0 = (unsigned)(uintptr_t)(&apanel[0]);
      const i32x4 z4 = {0, 0, 0, 0};
      tdm_load(tdm_group0(lds0, h + (size_t)(off + slot0) * HDIM + kc * 512, /*gather=*/false),
               tdm_group1(HDIM, NTOK * KTOP, 512, (unsigned)valid, HDIM),
               z4, z4);
      __builtin_amdgcn_s_wait_tensorcnt(0);
    }
#else
    for (int i = tid; i < ROWS * 64; i += 256) {
      const int r = i >> 6, c8 = i & 63;
      uint4 val = {0, 0, 0, 0};
      if (r < valid)
        val = *(const uint4*)(h + (size_t)(off + slot0 + r) * HDIM + kc * 512 + c8 * 8);
      *(uint4*)(&apanel[r * 512 + c8 * 8]) = val;
    }
#endif
    __syncthreads();
#pragma unroll
    for (int ks = 0; ks < 16; ++ks) {
      const size_t bblk = (size_t)(e * (DDIM / 16) + dt16) * (HDIM / 32) + (kc * 16 + ks);
      const v16bf b = load_b_frag(w2f, bblk, lane);
#pragma unroll
      for (int mt = 0; mt < 4; ++mt) {
        const v16bf a = load_a_frag(apanel + mt * 16 * 512, lane, ks);
        acc[mt] = __builtin_amdgcn_wmma_f32_16x16x32_bf16(false, a, false, b, (short)0,
                                                          acc[mt], false, false);
      }
    }
  }
  const int n = lane & 15;
  const int radd = (lane & 16) ? 8 : 0;
  const int dcol = dt16 * 16 + n;
  const float bias = b2[e * DDIM + dcol];
#pragma unroll
  for (int mt = 0; mt < 4; ++mt) {
#pragma unroll
    for (int r = 0; r < 8; ++r) {
      const int m = mt * 16 + r + radd;
      const int s = slot0 + m;
      if (s < c) {
        const float v = (acc[mt][r] + bias) * gws[m];
        contrib[((size_t)toks[m] * KTOP + kids[m]) * DDIM + dcol] = v;
      }
    }
  }
}

// ---------------------------------------------------------------------------
// Combine: out = x + contrib[slot0] + contrib[slot1] (fixed summation order).
// ---------------------------------------------------------------------------
__global__ __launch_bounds__(256) void combine_kernel(const float* __restrict__ x,
                                                      const float* __restrict__ contrib,
                                                      float* __restrict__ out) {
  const size_t i = ((size_t)blockIdx.x * 256 + threadIdx.x) * 4;
  const size_t n = i >> 10;      // / DDIM
  const size_t d = i & 1023;     // % DDIM
  float4 xv = *(const float4*)(x + i);
  float4 a  = *(const float4*)(contrib + (n * KTOP) * DDIM + d);
  float4 b  = *(const float4*)(contrib + (n * KTOP + 1) * DDIM + d);
  float4 o;
  o.x = xv.x + a.x + b.x;
  o.y = xv.y + a.y + b.y;
  o.z = xv.z + a.z + b.z;
  o.w = xv.w + a.w + b.w;
  *(float4*)(out + i) = o;
}

// ---------------------------------------------------------------------------
extern "C" void kernel_launch(void* const* d_in, const int* in_sizes, int n_in,
                              void* d_out, int out_size, void* d_ws, size_t ws_size,
                              hipStream_t stream) {
  const float* x        = (const float*)d_in[0];
  const float* ln_gamma = (const float*)d_in[1];
  const float* ln_beta  = (const float*)d_in[2];
  const float* gate_w   = (const float*)d_in[3];
  const float* w1       = (const float*)d_in[4];
  const float* b1       = (const float*)d_in[5];
  const float* w2       = (const float*)d_in[6];
  const float* b2       = (const float*)d_in[7];

  float* out       = (float*)d_out;
  float* loss_out  = out + (size_t)NTOK * DDIM;
  float* probs_out = loss_out + 1;

  // Workspace carve (~250 MB)
  char* p = (char*)d_ws;
  auto take = [&](size_t bytes) -> void* {
    void* q = (void*)p;
    p += (bytes + 255) & ~(size_t)255;
    return q;
  };
  float*  xn       = (float*)take((size_t)NTOK * DDIM * 4);
  __bf16* xnb      = (__bf16*)take((size_t)NTOK * DDIM * 2);
  __bf16* w1f      = (__bf16*)take((size_t)NEXP * DDIM * HDIM * 2);
  __bf16* w2f      = (__bf16*)take((size_t)NEXP * DDIM * HDIM * 2);
  __bf16* hbuf     = (__bf16*)take((size_t)NTOK * KTOP * HDIM * 2);
  float*  contrib  = (float*)take((size_t)NTOK * KTOP * DDIM * 4);
  int*    toklist  = (int*)take((size_t)NEXP * NTOK * 4);
  float*  gwlist   = (float*)take((size_t)NEXP * NTOK * 4);
  int*    klist    = (int*)take((size_t)NEXP * NTOK * 4);
  int*    cnt      = (int*)take(64);
  int*    offs     = (int*)take(64);
  float*  probsSum = (float*)take(64);

  hipMemsetAsync(cnt, 0, 64, stream);

  // Prepack weights to bf16 fragment layout (2,097,152 threads each).
  prepack_kernel<<<8192, 256, 0, stream>>>(w1, w1f, DDIM, HDIM);
  prepack_kernel<<<8192, 256, 0, stream>>>(w2, w2f, HDIM, DDIM);

  ln_kernel<<<NTOK, 256, 0, stream>>>(x, ln_gamma, ln_beta, xn, xnb);
  router_kernel<<<NTOK, 256, 0, stream>>>(xn, gate_w, probs_out, cnt, toklist, gwlist, klist);
  probs_reduce_kernel<<<NEXP, 256, 0, stream>>>(probs_out, probsSum);
  offsets_kernel<<<1, 1, 0, stream>>>(cnt, offs);

  ffn1_kernel<<<dim3(HDIM / 128, NTOK / ROWS, NEXP), 256, 0, stream>>>(
      xnb, w1f, b1, hbuf, toklist, cnt, offs);
  ffn2_kernel<<<dim3(DDIM / 128, NTOK / ROWS, NEXP), 256, 0, stream>>>(
      hbuf, w2f, b2, contrib, toklist, gwlist, klist, cnt, offs);

  combine_kernel<<<(NTOK * DDIM) / 1024, 256, 0, stream>>>(x, contrib, out);
  loss_kernel<<<1, 1, 0, stream>>>(cnt, probsSum, loss_out);
}
